// TransformerDecoderLayer_equi_21603685499621
// MI455X (gfx1250) — matrix operand order repaired and checked by reading, not compiled, not run on hardware
//
#include <hip/hip_runtime.h>
#include <hip/hip_bf16.h>
#include <cstdint>

// ---------------------------------------------------------------------------
// MI455X (gfx1250) VN-Transformer decoder layer.
// All channel-mixing einsums -> bf16 WMMA GEMMs (v_wmma_f32_16x16x32_bf16),
// double-buffered LDS staging via CDNA5 async-to-LDS + s_wait_asynccnt.
// N/K are compile-time so C-store addressing folds to immediate offsets.
// Attention / layernorm / VN-relu in fp32 VALU (tiny FLOP share).
// ---------------------------------------------------------------------------

typedef __attribute__((ext_vector_type(16))) __bf16 v16bf;
typedef __attribute__((ext_vector_type(8)))  __bf16 v8bf;
typedef __attribute__((ext_vector_type(8)))  float  v8f;

#define D_MODEL 128
#define NHEAD   8
#define HD      16
#define DFF     2048
#define NQ      2048
#define NKROWS  4096
#define LNEI    64
#define MQ      (NQ * 3)      // 6144
#define MK      (NKROWS * 3)  // 12288

// ------------------------------- GEMM --------------------------------------
#define BM 128
#define BN 128
#define BK 32
#define STR 40   // bf16 elems per LDS row: 32 + 8 pad (80B, 16B aligned)

__device__ __forceinline__ void async_b128(unsigned lds_off, const void* gaddr) {
  asm volatile("global_load_async_to_lds_b128 %0, %1, off"
               :: "v"(lds_off), "v"((unsigned long long)(uintptr_t)gaddr)
               : "memory");
}

// C[M,N] (f32 or bf16) = A[M,K](bf16) * W[N,K](bf16)^T ; M%128==0
template<int N, int K, bool OUT_BF16>
__global__ __launch_bounds__(256)
void gemm_bf16_wmma(const __bf16* __restrict__ A, const __bf16* __restrict__ W,
                    float* __restrict__ Cf, __bf16* __restrict__ Cb, int M) {
  __shared__ __align__(16) __bf16 smA[2][BM * STR];
  __shared__ __align__(16) __bf16 smB[2][BN * STR];

  const int tid  = threadIdx.x;
  const int lane = tid & 31;
  const int wave = tid >> 5;
  const int wm   = wave & 1;   // 2 waves along M (64 rows each)
  const int wn   = wave >> 1;  // 4 waves along N (32 cols each)
  const int blockM = blockIdx.x * BM;
  const int blockN = blockIdx.y * BN;

  v8f acc[4][2];
#pragma unroll
  for (int i = 0; i < 4; ++i)
#pragma unroll
    for (int j = 0; j < 2; ++j) {
      v8f z = {0.f, 0.f, 0.f, 0.f, 0.f, 0.f, 0.f, 0.f};
      acc[i][j] = z;
    }

  // per-thread staging chunk coordinates (512 x 16B chunks per tile, 2/thread)
  const int row0 = tid >> 2, cc0 = tid & 3;             // chunk tid
  const int row1 = (tid + 256) >> 2, cc1 = tid & 3;     // chunk tid+256
  const unsigned aoff0 = (unsigned)(uintptr_t)(&smA[0][row0 * STR + cc0 * 8]);
  const unsigned aoff1 = (unsigned)(uintptr_t)(&smA[0][row1 * STR + cc1 * 8]);
  const unsigned boff0 = (unsigned)(uintptr_t)(&smB[0][row0 * STR + cc0 * 8]);
  const unsigned boff1 = (unsigned)(uintptr_t)(&smB[0][row1 * STR + cc1 * 8]);
  const unsigned bufStride = (unsigned)(BM * STR * 2);  // bytes per buffer

  auto stage = [&](int kb, int buf) {
    const int k0 = kb * BK;
    const unsigned bb = buf * bufStride;
    async_b128(aoff0 + bb, A + (size_t)(blockM + row0) * K + k0 + cc0 * 8);
    async_b128(boff0 + bb, W + (size_t)(blockN + row0) * K + k0 + cc0 * 8);
    async_b128(aoff1 + bb, A + (size_t)(blockM + row1) * K + k0 + cc1 * 8);
    async_b128(boff1 + bb, W + (size_t)(blockN + row1) * K + k0 + cc1 * 8);
  };

  const int rsel  = lane & 15;
  const int koffA = (lane < 16) ? 0 : 8;   // A: K interleave {0..7,16..23}/{8..15,24..31}
  const int koffB = (lane < 16) ? 0 : 8;   // B: K interleave {0..7,16..23}/{8..15,24..31}

  const int nK = K / BK;
  stage(0, 0);
  for (int kb = 0; kb < nK; ++kb) {
    const int buf = kb & 1;
    if (kb + 1 < nK) {
      stage(kb + 1, buf ^ 1);
      // drain only stage kb's 4 async loads; stage kb+1 stays in flight
      asm volatile("s_wait_asynccnt 4" ::: "memory");
    } else {
      asm volatile("s_wait_asynccnt 0" ::: "memory");
    }
    __syncthreads();

    v16bf bfrag[2];
#pragma unroll
    for (int nt = 0; nt < 2; ++nt) {
      const __bf16* p = &smB[buf][(wn * 32 + nt * 16 + rsel) * STR + koffB];
      v8bf lo = *(const v8bf*)p;
      v8bf hi = *(const v8bf*)(p + 16);
      bfrag[nt] = __builtin_shufflevector(lo, hi, 0,1,2,3,4,5,6,7,8,9,10,11,12,13,14,15);
    }
#pragma unroll
    for (int mt = 0; mt < 4; ++mt) {
      const __bf16* p = &smA[buf][(wm * 64 + mt * 16 + rsel) * STR + koffA];
      v8bf lo = *(const v8bf*)p;
      v8bf hi = *(const v8bf*)(p + 16);
      v16bf afrag = __builtin_shufflevector(lo, hi, 0,1,2,3,4,5,6,7,8,9,10,11,12,13,14,15);
#pragma unroll
      for (int nt = 0; nt < 2; ++nt)
        acc[mt][nt] = __builtin_amdgcn_wmma_f32_16x16x32_bf16(
            false, afrag, false, bfrag[nt], (short)0, acc[mt][nt], false, false);
    }
    __syncthreads();  // protect buf from being overwritten by stage kb+2
  }

  // epilogue: C layout -> col = lane&15, row = vgpr + (lane>=16 ? 8 : 0)
  const int rextra = (lane < 16) ? 0 : 8;
#pragma unroll
  for (int mt = 0; mt < 4; ++mt)
#pragma unroll
    for (int nt = 0; nt < 2; ++nt) {
      const int col = blockN + wn * 32 + nt * 16 + (lane & 15);
      const size_t base =
          (size_t)(blockM + wm * 64 + mt * 16 + rextra) * N + col;
#pragma unroll
      for (int r = 0; r < 8; ++r) {
        if (OUT_BF16) Cb[base + (size_t)r * N] = (__bf16)acc[mt][nt][r];
        else          Cf[base + (size_t)r * N] = acc[mt][nt][r];
      }
    }
}

// --------------------------- helper kernels --------------------------------
__global__ void f32_to_bf16_k(const float* __restrict__ s, __bf16* __restrict__ d, int n) {
  int i = blockIdx.x * blockDim.x + threadIdx.x;
  if (i < n) d[i] = (__bf16)s[i];
}

// x[n, ch*3+c] (f32) -> A[(n*3+c)*128 + ch] (bf16)
__global__ void interleave_bf16_k(const float* __restrict__ x, __bf16* __restrict__ A, int total) {
  int i = blockIdx.x * blockDim.x + threadIdx.x;
  if (i >= total) return;
  int ch = i % D_MODEL;
  int m  = i / D_MODEL;
  int n  = m / 3, c = m % 3;
  A[i] = (__bf16)x[(size_t)n * (D_MODEL * 3) + ch * 3 + c];
}

// one block per query, one wave per head; 2 keys per lane (wave32)
__global__ __launch_bounds__(256)
void vn_attention_k(const float* __restrict__ Q, const float* __restrict__ Kf,
                    const float* __restrict__ Vf, const int* __restrict__ index_pair,
                    const int* __restrict__ key_cnt, const int* __restrict__ pair_batch,
                    __bf16* __restrict__ outb) {
  const int n = blockIdx.x;
  const int h = threadIdx.x >> 5;
  const int lane = threadIdx.x & 31;

  const int b = pair_batch[n];
  int kstart = 0;
  for (int j = 0; j < b; ++j) kstart += key_cnt[j];

  float q[48];
#pragma unroll
  for (int c = 0; c < 3; ++c)
#pragma unroll
    for (int dh = 0; dh < 16; ++dh)
      q[c * 16 + dh] = Q[(size_t)(n * 3 + c) * D_MODEL + h * HD + dh];

  float sc[2]; int g[2];
#pragma unroll
  for (int t = 0; t < 2; ++t) {
    const int l = lane + t * 32;
    const int idx = index_pair[(size_t)n * LNEI + l];
    const bool valid = idx >= 0;
    g[t] = kstart + (valid ? idx : 0);
    float s = 0.f;
#pragma unroll
    for (int c = 0; c < 3; ++c) {
      const float* kp = Kf + (size_t)(g[t] * 3 + c) * D_MODEL + h * HD;
#pragma unroll
      for (int dh = 0; dh < 16; ++dh) s += q[c * 16 + dh] * kp[dh];
    }
    sc[t] = valid ? s * 0.25f : -1e9f;  // 1/sqrt(16)
  }
  float m = fmaxf(sc[0], sc[1]);
#pragma unroll
  for (int o = 16; o >= 1; o >>= 1) m = fmaxf(m, __shfl_xor(m, o, 32));
  const float e0 = __expf(sc[0] - m), e1 = __expf(sc[1] - m);
  float s = e0 + e1;
#pragma unroll
  for (int o = 16; o >= 1; o >>= 1) s += __shfl_xor(s, o, 32);
  const float inv = 1.f / s;
  const float p0 = e0 * inv, p1 = e1 * inv;

#pragma unroll
  for (int c = 0; c < 3; ++c) {
    const float* v0 = Vf + (size_t)(g[0] * 3 + c) * D_MODEL + h * HD;
    const float* v1 = Vf + (size_t)(g[1] * 3 + c) * D_MODEL + h * HD;
#pragma unroll
    for (int dh = 0; dh < 16; ++dh) {
      float a = p0 * v0[dh] + p1 * v1[dh];
#pragma unroll
      for (int o = 16; o >= 1; o >>= 1) a += __shfl_xor(a, o, 32);
      if (lane == 0)
        outb[(size_t)(n * 3 + c) * D_MODEL + h * HD + dh] = (__bf16)a;
    }
  }
}

__device__ __forceinline__ void block_reduce_2(float& s1, float& s2, float* red) {
  const int lane = threadIdx.x & 31, wv = threadIdx.x >> 5;
#pragma unroll
  for (int o = 16; o >= 1; o >>= 1) { s1 += __shfl_xor(s1, o, 32); s2 += __shfl_xor(s2, o, 32); }
  if (lane == 0) { red[wv] = s1; red[4 + wv] = s2; }
  __syncthreads();
  s1 = red[0] + red[1] + red[2] + red[3];
  s2 = red[4] + red[5] + red[6] + red[7];
}

// residual + VN layernorm; xa in [n, ch*3+c] layout, xb in [m, ch] layout
template<bool WRITE_INTERLEAVED_OUT>
__global__ __launch_bounds__(128)
void vn_layernorm_k(const float* __restrict__ xa, const float* __restrict__ xb,
                    const float* __restrict__ gg, const float* __restrict__ bb,
                    float* __restrict__ hf, __bf16* __restrict__ hb,
                    float* __restrict__ out, int A_IS_INTERLEAVED) {
  __shared__ float red[8];
  const int n = blockIdx.x, ch = threadIdx.x;
  float y[3];
#pragma unroll
  for (int c = 0; c < 3; ++c) {
    float a = A_IS_INTERLEAVED ? xa[(size_t)n * (D_MODEL * 3) + ch * 3 + c]
                               : xa[(size_t)(n * 3 + c) * D_MODEL + ch];
    y[c] = a + xb[(size_t)(n * 3 + c) * D_MODEL + ch];
  }
  const float nr = sqrtf(y[0]*y[0] + y[1]*y[1] + y[2]*y[2] + 1e-6f);
  float s1 = nr, s2 = nr * nr;
  block_reduce_2(s1, s2, red);
  const float mu  = s1 * (1.f / D_MODEL);
  const float var = s2 * (1.f / D_MODEL) - mu * mu;
  const float nn  = (nr - mu) * rsqrtf(var + 1e-5f) * gg[ch] + bb[ch];
  const float scl = nn / nr;
#pragma unroll
  for (int c = 0; c < 3; ++c) {
    const float v = y[c] * scl;
    if (WRITE_INTERLEAVED_OUT) {
      out[(size_t)n * (D_MODEL * 3) + ch * 3 + c] = v;
    } else {
      hf[(size_t)(n * 3 + c) * D_MODEL + ch] = v;
      hb[(size_t)(n * 3 + c) * D_MODEL + ch] = (__bf16)v;
    }
  }
}

// VN relu: project out negative component along direction d (per (n, o))
__global__ void vn_relu_k(const __bf16* __restrict__ f, const __bf16* __restrict__ d,
                          __bf16* __restrict__ fp) {
  const int i = blockIdx.x * blockDim.x + threadIdx.x;  // over NQ*DFF
  const int o = i % DFF, n = i / DFF;
  float fv[3], dv[3];
#pragma unroll
  for (int c = 0; c < 3; ++c) {
    fv[c] = (float)f[(size_t)(n * 3 + c) * DFF + o];
    dv[c] = (float)d[(size_t)(n * 3 + c) * DFF + o];
  }
  const float dot = fv[0]*dv[0] + fv[1]*dv[1] + fv[2]*dv[2];
  const float dsq = dv[0]*dv[0] + dv[1]*dv[1] + dv[2]*dv[2] + 1e-6f;
  if (dot < 0.f) {
    const float t = dot / dsq;
#pragma unroll
    for (int c = 0; c < 3; ++c) fv[c] -= t * dv[c];
  }
#pragma unroll
  for (int c = 0; c < 3; ++c) fp[(size_t)(n * 3 + c) * DFF + o] = (__bf16)fv[c];
}

// ------------------------------ launch -------------------------------------
extern "C" void kernel_launch(void* const* d_in, const int* in_sizes, int n_in,
                              void* d_out, int out_size, void* d_ws, size_t ws_size,
                              hipStream_t stream) {
  (void)in_sizes; (void)n_in; (void)out_size; (void)ws_size;
  const float* tgt   = (const float*)d_in[0];
  const float* mem   = (const float*)d_in[1];
  const float* Wq    = (const float*)d_in[2];
  const float* Wk    = (const float*)d_in[3];
  const float* Wv    = (const float*)d_in[4];
  const float* Wo    = (const float*)d_in[5];
  const float* ln1g  = (const float*)d_in[6];
  const float* ln1b  = (const float*)d_in[7];
  const float* ln2g  = (const float*)d_in[8];
  const float* ln2b  = (const float*)d_in[9];
  const float* W1    = (const float*)d_in[10];
  const float* Wd    = (const float*)d_in[11];
  const float* W2    = (const float*)d_in[12];
  const int* index_pair = (const int*)d_in[13];
  const int* key_cnt    = (const int*)d_in[15];
  const int* pair_batch = (const int*)d_in[16];
  float* out = (float*)d_out;

  char* ws = (char*)d_ws;
  size_t cur = 0;
  auto alloc = [&](size_t bytes) -> void* {
    void* p = ws + cur;
    cur = (cur + bytes + 255) & ~(size_t)255;
    return p;
  };
  __bf16* wq_b  = (__bf16*)alloc((size_t)128 * 128 * 2);
  __bf16* wk_b  = (__bf16*)alloc((size_t)128 * 128 * 2);
  __bf16* wv_b  = (__bf16*)alloc((size_t)128 * 128 * 2);
  __bf16* wo_b  = (__bf16*)alloc((size_t)128 * 128 * 2);
  __bf16* w1_b  = (__bf16*)alloc((size_t)DFF * 128 * 2);
  __bf16* wd_b  = (__bf16*)alloc((size_t)DFF * DFF * 2);
  __bf16* w2_b  = (__bf16*)alloc((size_t)128 * DFF * 2);
  __bf16* a_tgt = (__bf16*)alloc((size_t)MQ * 128 * 2);
  __bf16* a_mem = (__bf16*)alloc((size_t)MK * 128 * 2);
  float*  qf    = (float*) alloc((size_t)MQ * 128 * 4);
  float*  kf    = (float*) alloc((size_t)MK * 128 * 4);
  float*  vf    = (float*) alloc((size_t)MK * 128 * 4);
  __bf16* ao_b  = (__bf16*)alloc((size_t)MQ * 128 * 2);
  float*  t2_f  = (float*) alloc((size_t)MQ * 128 * 4);
  float*  h_f   = (float*) alloc((size_t)MQ * 128 * 4);
  __bf16* h_b   = (__bf16*)alloc((size_t)MQ * 128 * 2);
  __bf16* f_b   = (__bf16*)alloc((size_t)MQ * DFF * 2);
  __bf16* dd_b  = (__bf16*)alloc((size_t)MQ * DFF * 2);
  __bf16* fp_b  = (__bf16*)alloc((size_t)MQ * DFF * 2);
  float*  f2_f  = (float*) alloc((size_t)MQ * 128 * 4);

  const dim3 t256(256);
  auto cvt = [&](const float* s, __bf16* d, int n) {
    f32_to_bf16_k<<<(n + 255) / 256, t256, 0, stream>>>(s, d, n);
  };
  cvt(Wq, wq_b, 128 * 128);   cvt(Wk, wk_b, 128 * 128);
  cvt(Wv, wv_b, 128 * 128);   cvt(Wo, wo_b, 128 * 128);
  cvt(W1, w1_b, DFF * 128);   cvt(Wd, wd_b, DFF * DFF);
  cvt(W2, w2_b, 128 * DFF);

  interleave_bf16_k<<<(MQ * 128) / 256, t256, 0, stream>>>(tgt, a_tgt, MQ * 128);
  interleave_bf16_k<<<(MK * 128) / 256, t256, 0, stream>>>(mem, a_mem, MK * 128);

  // q/k/v projections: C = A * W^T
  gemm_bf16_wmma<128, 128, false><<<dim3(MQ / BM, 1), t256, 0, stream>>>(a_tgt, wq_b, qf, nullptr, MQ);
  gemm_bf16_wmma<128, 128, false><<<dim3(MK / BM, 1), t256, 0, stream>>>(a_mem, wk_b, kf, nullptr, MK);
  gemm_bf16_wmma<128, 128, false><<<dim3(MK / BM, 1), t256, 0, stream>>>(a_mem, wv_b, vf, nullptr, MK);

  vn_attention_k<<<NQ, t256, 0, stream>>>(qf, kf, vf, index_pair, key_cnt, pair_batch, ao_b);

  // tgt2 = Wo @ attn_out
  gemm_bf16_wmma<128, 128, false><<<dim3(MQ / BM, 1), t256, 0, stream>>>(ao_b, wo_b, t2_f, nullptr, MQ);

  // h = VN-LN(tgt + tgt2)
  vn_layernorm_k<false><<<NQ, 128, 0, stream>>>(tgt, t2_f, ln1g, ln1b, h_f, h_b, nullptr, 1);

  // FFN: f = W1 h ; d = Wd f ; f' = VNReLU(f, d) ; f2 = W2 f'
  gemm_bf16_wmma<DFF, 128, true><<<dim3(MQ / BM, DFF / BN), t256, 0, stream>>>(h_b, w1_b, nullptr, f_b, MQ);
  gemm_bf16_wmma<DFF, DFF, true><<<dim3(MQ / BM, DFF / BN), t256, 0, stream>>>(f_b, wd_b, nullptr, dd_b, MQ);
  vn_relu_k<<<(NQ * DFF) / 256, t256, 0, stream>>>(f_b, dd_b, fp_b);
  gemm_bf16_wmma<128, DFF, false><<<dim3(MQ / BM, 1), t256, 0, stream>>>(fp_b, w2_b, f2_f, nullptr, MQ);

  // out = VN-LN(h + f2), written back in reference [n, ch*3+c] layout
  vn_layernorm_k<true><<<NQ, 128, 0, stream>>>(h_f, f2_f, ln2g, ln2b, nullptr, nullptr, out, 0);
}
